// SpatialSSM_3015067041908
// MI455X (gfx1250) — matrix operand order repaired and checked by reading, not compiled, not run on hardware
//
#include <hip/hip_runtime.h>
#include <stddef.h>
#include <stdint.h>

// ---------------------------------------------------------------------------
// Mamba-2-layer forward for MI455X (gfx1250, wave32).
//  * GEMMs: v_wmma_f32_16x16x32_bf16, bf16 hi/lo split (3 WMMAs per K-chunk)
//    for ~fp32 accuracy at bf16-WMMA rate. Per-wave 32x64 register tile,
//    B fragments streamed per N-subtile to keep VGPR pressure spill-free.
//  * Scan: 512-step recurrence, per-step B/C vectors double-buffered in LDS
//    via global_load_async_to_lds_b128 + s_wait_asynccnt (CDNA5 async path).
// ---------------------------------------------------------------------------

#define D_INPUT  512
#define D_MODEL  1024
#define D_STATE  16
#define D_CONV   4
#define D_INNER  2048
#define DT_RANK  64
#define N_LAYERS 2
#define BATCH    8
#define SEQ      512
#define BL       (BATCH * SEQ)   // 4096 rows in every GEMM

typedef __attribute__((ext_vector_type(16))) __bf16          v16bf;
typedef __attribute__((ext_vector_type(8)))  float           v8f;
typedef __attribute__((ext_vector_type(8)))  unsigned short  v8us;
typedef __attribute__((ext_vector_type(16))) unsigned short  v16us;

union Frag16 { v16bf bf; v16us us16; v8us us8[2]; };

// ----------------------------- helpers -------------------------------------

__device__ __forceinline__ unsigned short f32_to_bf16_rn(float f) {
  unsigned int u = __float_as_uint(f);
  unsigned int lsb = (u >> 16) & 1u;
  u += 0x7fffu + lsb;                 // round-to-nearest-even
  return (unsigned short)(u >> 16);
}
__device__ __forceinline__ float bf16_to_f32(unsigned short h) {
  return __uint_as_float(((unsigned int)h) << 16);
}

// fp32 -> (hi, lo) bf16 split; src may be a strided sub-matrix (ld >= cols).
__global__ void k_split(const float* __restrict__ src, int ld, int cols,
                        unsigned short* __restrict__ hi,
                        unsigned short* __restrict__ lo, int n) {
  int idx = blockIdx.x * blockDim.x + threadIdx.x;
  if (idx >= n) return;
  int r = idx / cols, c = idx - r * cols;
  float v = src[(size_t)r * ld + c];
  unsigned short h = f32_to_bf16_rn(v);
  float residual = v - bf16_to_f32(h);
  hi[idx] = h;
  lo[idx] = f32_to_bf16_rn(residual);
}

// A fragment (16x32 bf16, M x K): lane l holds row m=l&15.
// VGPR0..3 : K = koff..koff+7 (koff=(l>>4)*8); VGPR4..7 : K = 16+koff..16+koff+7
__device__ __forceinline__ v16bf load_a_frag(const unsigned short* __restrict__ A,
                                             int lda, int row0, int k0, int lane) {
  int m = lane & 15, sel = lane >> 4;
  const unsigned short* p = A + (size_t)(row0 + m) * lda + k0 + sel * 8;
  Frag16 f;
  f.us8[0] = *(const v8us*)(p);
  f.us8[1] = *(const v8us*)(p + 16);
  return f.bf;
}

// B fragment (32x16 bf16, K x N) from row-major W (N x K):
// lane l holds column n=l&15; VGPR0..3 : K = koff..koff+7 (koff=(l>>4)*8); VGPR4..7 : K = 16+koff..16+koff+7
__device__ __forceinline__ v16bf load_b_frag(const unsigned short* __restrict__ W,
                                             int ldw, int col0, int k0, int lane) {
  int n = lane & 15, sel = lane >> 4;
  const unsigned short* p = W + (size_t)(col0 + n) * ldw + k0 + sel * 8;
  Frag16 f;
  f.us8[0] = *(const v8us*)(p);
  f.us8[1] = *(const v8us*)(p + 16);
  return f.bf;
}

#define WMMA_BF16(A, B, Cacc) \
  __builtin_amdgcn_wmma_f32_16x16x32_bf16(false, (A), false, (B), (short)0, (Cacc), false, false)

// ------------------------- GEMM, 32x64 wave tile ---------------------------
// C[M,N] = A[M,K] * W[N,K]^T (+bias) (+=C if addC). Requires M%32==0, N%64==0.
// launch_bounds(256,1): allow full VGPR allocation, no accumulator spills.
__global__ void __launch_bounds__(256, 1)
k_gemm_bf16x3_32x64(const unsigned short* __restrict__ Ah,
                    const unsigned short* __restrict__ Al,
                    const unsigned short* __restrict__ Wh,
                    const unsigned short* __restrict__ Wl,
                    float* __restrict__ C,
                    const float* __restrict__ bias,
                    int M, int N, int K, int addC) {
  const int wave = threadIdx.x >> 5;
  const int lane = threadIdx.x & 31;
  const int tiles_n = N >> 6;
  const int tile = blockIdx.x * 8 + wave;
  if (tile >= (M >> 5) * tiles_n) return;
  const int tm = (tile / tiles_n) << 5;
  const int tn = (tile % tiles_n) << 6;

  v8f acc[2][4];
  #pragma unroll
  for (int i = 0; i < 2; ++i)
    #pragma unroll
    for (int j = 0; j < 4; ++j) acc[i][j] = (v8f){};

  for (int k0 = 0; k0 < K; k0 += 32) {
    v16bf a_h[2], a_l[2];
    #pragma unroll
    for (int i = 0; i < 2; ++i) {
      a_h[i] = load_a_frag(Ah, K, tm + 16 * i, k0, lane);
      a_l[i] = load_a_frag(Al, K, tm + 16 * i, k0, lane);
    }
    // Stream B fragments one N-subtile at a time (limits live VGPRs);
    // WMMA order per j: (i0,i1)*hh, (i0,i1)*hl, (i0,i1)*lh -> no back-to-back
    // dependent accumulators.
    #pragma unroll
    for (int j = 0; j < 4; ++j) {
      v16bf bh = load_b_frag(Wh, K, tn + 16 * j, k0, lane);
      v16bf bl = load_b_frag(Wl, K, tn + 16 * j, k0, lane);
      #pragma unroll
      for (int i = 0; i < 2; ++i) acc[i][j] = WMMA_BF16(a_h[i], bh, acc[i][j]);
      #pragma unroll
      for (int i = 0; i < 2; ++i) acc[i][j] = WMMA_BF16(a_h[i], bl, acc[i][j]);
      #pragma unroll
      for (int i = 0; i < 2; ++i) acc[i][j] = WMMA_BF16(a_l[i], bh, acc[i][j]);
    }
  }

  const int n = lane & 15, half = lane >> 4;
  #pragma unroll
  for (int i = 0; i < 2; ++i)
    #pragma unroll
    for (int j = 0; j < 4; ++j) {
      float bb = bias ? bias[tn + 16 * j + n] : 0.0f;
      #pragma unroll
      for (int r = 0; r < 8; ++r) {
        int m = tm + 16 * i + half * 8 + r;
        size_t idx = (size_t)m * N + tn + 16 * j + n;
        float v = acc[i][j][r] + bb;
        if (addC) v += C[idx];
        C[idx] = v;
      }
    }
}

// ------------------------- GEMM, 16x32 wave tile (small N) -----------------
__global__ void k_gemm_bf16x3_16x32(const unsigned short* __restrict__ Ah,
                                    const unsigned short* __restrict__ Al,
                                    const unsigned short* __restrict__ Wh,
                                    const unsigned short* __restrict__ Wl,
                                    float* __restrict__ C,
                                    const float* __restrict__ bias,
                                    int M, int N, int K, int addC) {
  const int wave = threadIdx.x >> 5;
  const int lane = threadIdx.x & 31;
  const int tiles_n = N >> 5;
  const int tile = blockIdx.x * 8 + wave;
  if (tile >= (M >> 4) * tiles_n) return;
  const int tm = (tile / tiles_n) << 4;
  const int tn = (tile % tiles_n) << 5;

  v8f acc0 = {}, acc1 = {};
  for (int k0 = 0; k0 < K; k0 += 32) {
    v16bf ah = load_a_frag(Ah, K, tm, k0, lane);
    v16bf al = load_a_frag(Al, K, tm, k0, lane);
    v16bf bh0 = load_b_frag(Wh, K, tn,      k0, lane);
    v16bf bl0 = load_b_frag(Wl, K, tn,      k0, lane);
    v16bf bh1 = load_b_frag(Wh, K, tn + 16, k0, lane);
    v16bf bl1 = load_b_frag(Wl, K, tn + 16, k0, lane);
    acc0 = WMMA_BF16(ah, bh0, acc0);
    acc1 = WMMA_BF16(ah, bh1, acc1);
    acc0 = WMMA_BF16(ah, bl0, acc0);
    acc1 = WMMA_BF16(ah, bl1, acc1);
    acc0 = WMMA_BF16(al, bh0, acc0);
    acc1 = WMMA_BF16(al, bh1, acc1);
  }

  const int n = lane & 15, half = lane >> 4;
  float b0 = bias ? bias[tn + n]      : 0.0f;
  float b1 = bias ? bias[tn + 16 + n] : 0.0f;
  #pragma unroll
  for (int r = 0; r < 8; ++r) {
    int m = tm + half * 8 + r;
    size_t i0 = (size_t)m * N + tn + n;
    size_t i1 = i0 + 16;
    float v0 = acc0[r] + b0;
    float v1 = acc1[r] + b1;
    if (addC) { v0 += C[i0]; v1 += C[i1]; }
    C[i0] = v0;
    C[i1] = v1;
  }
}

// ----------------------- depthwise causal conv + SiLU ----------------------
__global__ void k_conv_silu(const float* __restrict__ xz,     // (BL, 2*D_INNER)
                            const float* __restrict__ cw,     // (D_INNER, 4)
                            const float* __restrict__ cb,     // (D_INNER,)
                            float* __restrict__ xi_act) {     // (BL, D_INNER)
  int idx = blockIdx.x * blockDim.x + threadIdx.x;
  if (idx >= BL * D_INNER) return;
  int d = idx % D_INNER;
  int l = (idx / D_INNER) % SEQ;
  int b = idx / (D_INNER * SEQ);
  float acc = cb[d];
  #pragma unroll
  for (int j = 0; j < D_CONV; ++j) {
    int ls = l - (D_CONV - 1) + j;
    if (ls >= 0)
      acc += cw[d * D_CONV + j] * xz[((size_t)(b * SEQ + ls)) * (2 * D_INNER) + d];
  }
  xi_act[idx] = acc / (1.0f + expf(-acc));   // SiLU
}

// --------------------------- selective scan --------------------------------
// grid (BATCH, D_INNER/256), block 256. Per-step B_t/C_t (32 floats = 128B)
// double-buffered in LDS with async-to-LDS prefetch of step t+1.
__global__ void k_scan(const float* __restrict__ dpre,    // (BL, D_INNER) pre-softplus
                       const float* __restrict__ xi_act,  // (BL, D_INNER)
                       const float* __restrict__ xdbc,    // (BL, 96): [dt | B | C]
                       const float* __restrict__ xz,      // (BL, 2*D_INNER)
                       const float* __restrict__ A_log,   // (D_INNER, 16)
                       const float* __restrict__ Dp,      // (D_INNER,)
                       float* __restrict__ ycomb) {       // (BL, D_INNER)
  const int b = blockIdx.x;
  const int d = blockIdx.y * 256 + threadIdx.x;
  __shared__ float BC[2][2 * D_STATE];   // [buf][ B(16) | C(16) ], 128B each

  float a[D_STATE], hst[D_STATE];
  #pragma unroll
  for (int s = 0; s < D_STATE; ++s) {
    a[s] = -expf(A_log[(size_t)d * D_STATE + s]);
    hst[s] = 0.0f;
  }
  const float Dd = Dp[d];

  // Async prefetch of one step's B|C (128B) into LDS buffer `buf`:
  // lanes 0..7 of wave 0 each move one 16B chunk.
  auto prefetch = [&](int t, int buf) {
    if (threadIdx.x < 8) {
      const float* p = xdbc + ((size_t)(b * SEQ + t)) * 96 + DT_RANK;
      uint64_t gaddr = (uint64_t)(uintptr_t)p + (uint64_t)threadIdx.x * 16u;
      unsigned lds   = (unsigned)(uintptr_t)(&BC[buf][0]) + threadIdx.x * 16u;
      asm volatile("global_load_async_to_lds_b128 %0, %1, off"
                   :: "v"(lds), "v"(gaddr) : "memory");
    }
  };

  prefetch(0, 0);

  for (int t = 0; t < SEQ; ++t) {
    const int buf = t & 1;
    asm volatile("s_wait_asynccnt 0x0" ::: "memory");
    __syncthreads();                     // buffer `buf` ready; t-1 reads done
    if (t + 1 < SEQ) prefetch(t + 1, buf ^ 1);

    const size_t row = (size_t)(b * SEQ + t);
    float dp = dpre[row * D_INNER + d];
    float delta = (dp > 20.0f) ? dp : log1pf(expf(dp));   // softplus
    float u  = xi_act[row * D_INNER + d];
    float du = delta * u;
    float y = 0.0f;
    #pragma unroll
    for (int s = 0; s < D_STATE; ++s) {
      hst[s] = expf(delta * a[s]) * hst[s] + du * BC[buf][s];
      y += hst[s] * BC[buf][D_STATE + s];
    }
    float z = xz[row * (2 * D_INNER) + D_INNER + d];
    float yv = (y + u * Dd) * (z / (1.0f + expf(-z)));    // gate with SiLU(z)
    ycomb[row * D_INNER + d] = yv;
  }
}

// ----------------------- layernorm of last token ---------------------------
__global__ void k_ln_last(const float* __restrict__ h, const float* __restrict__ g,
                          const float* __restrict__ beta, float* __restrict__ out) {
  __shared__ float red[256];
  const int b = blockIdx.x, tid = threadIdx.x;
  const float* row = h + ((size_t)(b * SEQ + (SEQ - 1))) * D_MODEL;

  float s = 0.0f;
  for (int i = tid; i < D_MODEL; i += 256) s += row[i];
  red[tid] = s; __syncthreads();
  for (int o = 128; o; o >>= 1) { if (tid < o) red[tid] += red[tid + o]; __syncthreads(); }
  float mu = red[0] / (float)D_MODEL;
  __syncthreads();

  float v = 0.0f;
  for (int i = tid; i < D_MODEL; i += 256) { float dlt = row[i] - mu; v += dlt * dlt; }
  red[tid] = v; __syncthreads();
  for (int o = 128; o; o >>= 1) { if (tid < o) red[tid] += red[tid + o]; __syncthreads(); }
  float inv = rsqrtf(red[0] / (float)D_MODEL + 1e-5f);

  for (int i = tid; i < D_MODEL; i += 256)
    out[(size_t)b * D_MODEL + i] = (row[i] - mu) * inv * g[i] + beta[i];
}

// ------------------------------- host --------------------------------------

static inline size_t align_up(size_t v, size_t a) { return (v + a - 1) & ~(a - 1); }

extern "C" void kernel_launch(void* const* d_in, const int* in_sizes, int n_in,
                              void* d_out, int out_size, void* d_ws, size_t ws_size,
                              hipStream_t stream) {
  (void)in_sizes; (void)n_in; (void)out_size; (void)ws_size;
  const float* x      = (const float*)d_in[0];
  const float* proj_w = (const float*)d_in[1];
  const float* proj_b = (const float*)d_in[2];
  const float* W_in   = (const float*)d_in[3];
  const float* conv_w = (const float*)d_in[4];
  const float* conv_b = (const float*)d_in[5];
  const float* W_x    = (const float*)d_in[6];
  const float* W_dt   = (const float*)d_in[7];
  const float* b_dt   = (const float*)d_in[8];
  const float* A_log  = (const float*)d_in[9];
  const float* Dvec   = (const float*)d_in[10];
  const float* W_out  = (const float*)d_in[11];
  const float* ln_g   = (const float*)d_in[12];
  const float* ln_b   = (const float*)d_in[13];

  char* ws = (char*)d_ws;
  size_t off = 0;
  auto alloc = [&](size_t bytes) -> void* {
    void* p = ws + off;
    off = align_up(off + bytes, 256);
    return p;
  };

  float* h     = (float*)alloc((size_t)BL * D_MODEL * 4);
  float* xz    = (float*)alloc((size_t)BL * 2 * D_INNER * 4);
  float* xia   = (float*)alloc((size_t)BL * D_INNER * 4);
  float* xdbc  = (float*)alloc((size_t)BL * 96 * 4);
  float* dpre  = (float*)alloc((size_t)BL * D_INNER * 4);
  float* yc    = (float*)alloc((size_t)BL * D_INNER * 4);
  unsigned short* ah = (unsigned short*)alloc((size_t)BL * D_INNER * 2);
  unsigned short* al = (unsigned short*)alloc((size_t)BL * D_INNER * 2);
  unsigned short* wh = (unsigned short*)alloc((size_t)2 * D_INNER * D_MODEL * 2);
  unsigned short* wl = (unsigned short*)alloc((size_t)2 * D_INNER * D_MODEL * 2);

  auto split = [&](const float* src, int rows, int cols, int ld,
                   unsigned short* hi, unsigned short* lo) {
    int n = rows * cols;
    k_split<<<(n + 255) / 256, 256, 0, stream>>>(src, ld, cols, hi, lo, n);
  };
  auto gemm = [&](int M, int N, int K, float* C, const float* bias, int addC) {
    if ((M & 31) == 0 && (N & 63) == 0) {
      int tiles = (M >> 5) * (N >> 6);
      k_gemm_bf16x3_32x64<<<(tiles + 7) / 8, 256, 0, stream>>>(ah, al, wh, wl, C, bias, M, N, K, addC);
    } else {
      int tiles = (M >> 4) * (N >> 5);
      k_gemm_bf16x3_16x32<<<(tiles + 7) / 8, 256, 0, stream>>>(ah, al, wh, wl, C, bias, M, N, K, addC);
    }
  };

  // h = x @ proj_w^T + proj_b
  split(x, BL, D_INPUT, D_INPUT, ah, al);
  split(proj_w, D_MODEL, D_INPUT, D_INPUT, wh, wl);
  gemm(BL, D_MODEL, D_INPUT, h, proj_b, 0);

  for (int l = 0; l < N_LAYERS; ++l) {
    // xz = h @ W_in^T
    split(h, BL, D_MODEL, D_MODEL, ah, al);
    split(W_in + (size_t)l * 2 * D_INNER * D_MODEL, 2 * D_INNER, D_MODEL, D_MODEL, wh, wl);
    gemm(BL, 2 * D_INNER, D_MODEL, xz, nullptr, 0);

    // xi = silu(causal_conv(xi) + conv_b)
    k_conv_silu<<<(BL * D_INNER + 255) / 256, 256, 0, stream>>>(
        xz, conv_w + (size_t)l * D_INNER * D_CONV, conv_b + (size_t)l * D_INNER, xia);

    // xdbc = xi @ W_x^T  (96 = dt(64) | B(16) | C(16))
    split(xia, BL, D_INNER, D_INNER, ah, al);
    split(W_x + (size_t)l * 96 * D_INNER, 96, D_INNER, D_INNER, wh, wl);
    gemm(BL, 96, D_INNER, xdbc, nullptr, 0);

    // delta_pre = dt @ W_dt^T + b_dt  (softplus fused in scan)
    split(xdbc, BL, DT_RANK, 96, ah, al);                       // strided dt slice
    split(W_dt + (size_t)l * D_INNER * DT_RANK, D_INNER, DT_RANK, DT_RANK, wh, wl);
    gemm(BL, D_INNER, DT_RANK, dpre, b_dt + (size_t)l * D_INNER, 0);

    // sequential selective scan + u*D + SiLU(z) gate
    k_scan<<<dim3(BATCH, D_INNER / 256), 256, 0, stream>>>(
        dpre, xia, xdbc, xz,
        A_log + (size_t)l * D_INNER * D_STATE, Dvec + (size_t)l * D_INNER, yc);

    // h += ycomb @ W_out^T
    split(yc, BL, D_INNER, D_INNER, ah, al);
    split(W_out + (size_t)l * D_MODEL * D_INNER, D_MODEL, D_INNER, D_INNER, wh, wl);
    gemm(BL, D_MODEL, D_INNER, h, nullptr, 1);
  }

  // layernorm, emit last token only
  k_ln_last<<<BATCH, 256, 0, stream>>>(h, ln_g, ln_b, (float*)d_out);
}
